// Residual_Norm_88587995447807
// MI455X (gfx1250) — matrix operand order repaired and checked by reading, not compiled, not run on hardware
//
#include <hip/hip_runtime.h>
#include <stdint.h>

typedef __attribute__((ext_vector_type(16))) _Float16     v16h;
typedef __attribute__((ext_vector_type(8)))  float        v8f;
typedef __attribute__((ext_vector_type(4)))  unsigned int u32x4;
typedef __attribute__((ext_vector_type(8)))  int          i32x8;
typedef __attribute__((ext_vector_type(4)))  int          i32x4;

#define LDS_STRIDE 40   // halfs per row (32 data + 8 pad) -> conflict-free b128 reads

// TDM available? (ROCm 7.2 clang-22: 5-arg builtin; amdgpu-toolchain clang-23: 6-arg)
#if defined(__AMDGCN__) && defined(__has_builtin)
#  if __has_builtin(__builtin_amdgcn_tensor_load_to_lds)
#    define USE_TDM 1
#  endif
#endif
#ifndef USE_TDM
#  define USE_TDM 0
#endif

// ---------------------------------------------------------------- utilities
__device__ __forceinline__ float wave_sum(float v) {
  #pragma unroll
  for (int m = 16; m > 0; m >>= 1) v += __shfl_xor(v, m, 32);
  return v;
}

__global__ void zero_ws_kernel(unsigned int* __restrict__ p, size_t n) {
  size_t stride = (size_t)gridDim.x * blockDim.x;
  for (size_t i = (size_t)blockIdx.x * blockDim.x + threadIdx.x; i < n; i += stride)
    p[i] = 0u;
}

// ------------------------------------------------- weight / bias repacking
// QKV weights (3x 1024x128x3x3 f32) -> (3072, 1152) f16 with k = q*128 + c
__global__ void pack_qkv_kernel(const float* __restrict__ Wq, const float* __restrict__ Wk,
                                const float* __restrict__ Wv, const float* __restrict__ bq,
                                const float* __restrict__ bk, const float* __restrict__ bv,
                                _Float16* __restrict__ wOut, float* __restrict__ bOut) {
  size_t gid = (size_t)blockIdx.x * blockDim.x + threadIdx.x;
  const size_t total = (size_t)3072 * 1152;
  if (gid < total) {
    int m = (int)(gid / 1152);
    int k = (int)(gid % 1152);
    int q = k >> 7, c = k & 127;
    const float* W = (m < 1024) ? Wq : ((m < 2048) ? Wk : Wv);
    int o = m & 1023;
    wOut[gid] = (_Float16)W[(size_t)o * 1152 + (size_t)c * 9 + q];
  }
  if (gid < 3072) {
    int m = (int)gid;
    bOut[m] = (m < 1024) ? bq[m] : ((m < 2048) ? bk[m - 1024] : bv[m - 2048]);
  }
}

// Wp (128x1024x3x3 f32) -> (128, 9216) f16 with k = q*1024 + c
__global__ void pack_p_kernel(const float* __restrict__ Wp, _Float16* __restrict__ wOut) {
  size_t gid = (size_t)blockIdx.x * blockDim.x + threadIdx.x;
  const size_t total = (size_t)128 * 9216;
  if (gid >= total) return;
  int o = (int)(gid / 9216);
  int k = (int)(gid % 9216);
  int q = k >> 10, c = k & 1023;
  wOut[gid] = (_Float16)Wp[(size_t)o * 9216 + (size_t)c * 9 + q];
}

// ---------------------------------------------------------- GroupNorm(1) stats
__global__ __launch_bounds__(256) void gn_stats_kernel(const float* __restrict__ x,
                                                       float* __restrict__ stats) {
  int n = blockIdx.y;
  const float* p = x + (size_t)n * 524288 + (size_t)blockIdx.x * 8192;
  int tid = threadIdx.x, lane = tid & 31, wv = tid >> 5;
  float s = 0.f, q = 0.f;
  #pragma unroll
  for (int i = 0; i < 32; ++i) {
    float v = p[tid + i * 256];
    s += v; q += v * v;
  }
  s = wave_sum(s); q = wave_sum(q);
  __shared__ float sS[8], sQ[8];
  if (lane == 0) { sS[wv] = s; sQ[wv] = q; }
  __syncthreads();
  if (tid == 0) {
    float a = 0.f, b = 0.f;
    #pragma unroll
    for (int i = 0; i < 8; ++i) { a += sS[i]; b += sQ[i]; }
    atomicAdd(&stats[2 * n + 0], a);
    atomicAdd(&stats[2 * n + 1], b);
  }
}

// normalize x and write f16 into padded (N,128,66,66) buffer (borders pre-zeroed)
__global__ void gn_apply_pad_kernel(const float* __restrict__ x, const float* __restrict__ stats,
                                    const float* __restrict__ g, const float* __restrict__ b,
                                    _Float16* __restrict__ pad) {
  size_t gid = (size_t)blockIdx.x * blockDim.x + threadIdx.x;
  if (gid >= (size_t)8 * 128 * 4096) return;
  int n = (int)(gid >> 19);
  int c = (int)((gid >> 12) & 127);
  int l = (int)(gid & 4095);
  float mean = stats[2 * n + 0] * (1.f / 524288.f);
  float var  = stats[2 * n + 1] * (1.f / 524288.f) - mean * mean;
  float rstd = rsqrtf(var + 1e-5f);
  float v = (x[gid] - mean) * rstd * g[c] + b[c];
  size_t dst = (((size_t)(n * 128 + c) * 66) + (l >> 6) + 1) * 66 + (l & 63) + 1;
  pad[dst] = (_Float16)v;
}

// ------------------------------------------------------- implicit-GEMM conv3x3
// inPad : (Nimg, Cin, 66, 66) f16   (zero-padded, normalized)
// w     : (Cout, Kdim) f16, Kdim = 9*Cin, k = q*Cin + c, q = 3*r + s
// out   : (Nimg, Cout, 4096) f16 (F16OUT) or f32
// grid  : (Cout/128, 32, Nimg), block 256 (8 waves, 2x4 wave tile of 64x32)
template <bool F16OUT>
__global__ __launch_bounds__(256) void conv3x3_wmma_kernel(
    const _Float16* __restrict__ inPad, const _Float16* __restrict__ w,
    const float* __restrict__ bias, _Float16* __restrict__ outH,
    float* __restrict__ outF, int Cin, int Cout, int Kdim) {
  __shared__ __align__(16) _Float16 Al[2][128 * LDS_STRIDE];  // [m][k]
  __shared__ __align__(16) _Float16 Bl[2][128 * LDS_STRIDE];  // [n][k] (transposed)

  const int tid  = threadIdx.x;
  const int lane = tid & 31;
  const int wv   = tid >> 5;
  const int wm   = wv & 1;       // 0..1  -> 64-row M slab
  const int wn   = wv >> 1;      // 0..3  -> 32-col N slab
  const int img  = blockIdx.z;
  const int cout0 = blockIdx.x * 128;
  const int p0    = blockIdx.y * 128;
  const int KT    = Kdim >> 5;

#if USE_TDM
  // Tensor Data Mover: DMA the 128x32-half weight tile into LDS, with
  // 4-DWORD padding after every 16 DWORDs -> reproduces LDS_STRIDE=40 halfs.
  auto tdmA = [&](int kt, int buf) {
    const int k0 = kt << 5;
    unsigned lds = (unsigned)(uintptr_t)&Al[buf][0];
    unsigned long long ga =
        (unsigned long long)(uintptr_t)(w + (size_t)cout0 * Kdim + k0);
    u32x4 g0;
    g0.x = 1u;                                           // count=1 (valid D#)
    g0.y = lds;                                          // lds_addr
    g0.z = (unsigned)ga;                                 // global_addr[31:0]
    g0.w = (unsigned)((ga >> 32) & 0x01FFFFFFull) | 0x80000000u;  // addr[56:32] | type=2
    i32x8 g1;
    g1[0] = (1 << 16)      // data_size = 1 -> 2 bytes
          | (1 << 20)      // pad_enable
          | (3 << 22)      // pad_interval = 3 -> every 16 DWORDs (32 halfs)
          | (3 << 25);     // pad_amount   = 3 -> 4 DWORDs (8 halfs)
    g1[1] = (32 << 16);    // tensor_dim0[15:0]  = 32 (halfs per row)
    g1[2] = (128 << 16);   // tensor_dim0 hi = 0 | tensor_dim1[15:0] = 128 rows
    g1[3] = (32 << 16);    // tensor_dim1 hi = 0 | tile_dim0 = 32
    g1[4] = 128;           // tile_dim1 = 128 | tile_dim2 = 0
    g1[5] = Kdim;          // tensor_dim0_stride[31:0] = Kdim (halfs)
    g1[6] = 0;             // stride hi | tensor_dim1_stride lo
    g1[7] = 0;
    i32x4 z4; z4[0] = 0; z4[1] = 0; z4[2] = 0; z4[3] = 0;  // groups 2/3: <=2D tensor
#if defined(__clang_major__) && (__clang_major__ >= 23)
    i32x8 z8;
    #pragma unroll
    for (int i = 0; i < 8; ++i) z8[i] = 0;
    __builtin_amdgcn_tensor_load_to_lds(g0, g1, z4, z4, z8, 0);  // 6-arg (clang-23)
#else
    __builtin_amdgcn_tensor_load_to_lds(g0, g1, z4, z4, 0);      // 5-arg (ROCm 7.2)
#endif
  };
#endif

  auto stageA = [&](int kt, int buf) {   // manual fallback path
    const int k0   = kt << 5;
    const int row  = tid >> 1;
    const int part = (tid & 1) << 4;  // 0 or 16 halfs
    const uint4* g = (const uint4*)(w + (size_t)(cout0 + row) * Kdim + k0 + part);
    uint4 x0 = g[0], x1 = g[1];
    uint4* d = (uint4*)(&Al[buf][row * LDS_STRIDE + part]);
    d[0] = x0; d[1] = x1;
  };
  auto stageB = [&](int kt, int buf) {
    const int k0    = kt << 5;
    const int q     = k0 / Cin;         // Cin is a multiple of 32 -> whole tile shares q
    const int cbase = k0 - q * Cin;
    const int r = q / 3, s = q - 3 * r;
    const int kk  = tid >> 3;           // 0..31 local k
    const int c   = cbase + kk;
    const int nn0 = (tid & 7) << 4;     // 16 pixels, never straddles an image row
    const int l0  = p0 + nn0;
    const int hi  = l0 >> 6, wi0 = l0 & 63;
    const _Float16* src = inPad + (((size_t)(img * Cin + c) * 66) + hi + r) * 66 + wi0 + s;
    _Float16* dst = &Bl[buf][0] + kk;
    #pragma unroll
    for (int j = 0; j < 16; ++j)
      dst[(size_t)(nn0 + j) * LDS_STRIDE] = src[j];
  };

  v8f acc[4][2];
  #pragma unroll
  for (int i = 0; i < 4; ++i)
    #pragma unroll
    for (int j = 0; j < 2; ++j)
      #pragma unroll
      for (int r = 0; r < 8; ++r) acc[i][j][r] = 0.f;

  const int mrow = lane & 15;
  const int kbA  = (lane < 16) ? 0 : 8;
  const int kbB  = (lane < 16) ? 0 : 8;

  auto compute = [&](int buf) {
    v16h a[4], b[2];
    #pragma unroll
    for (int i = 0; i < 4; ++i) {
      const uint4* p = (const uint4*)(&Al[buf][(wm * 64 + i * 16 + mrow) * LDS_STRIDE + kbA]);
      ((uint4*)&a[i])[0] = p[0];   // K = kb .. kb+7
      ((uint4*)&a[i])[1] = p[2];   // K = kb+16 .. kb+23
    }
    #pragma unroll
    for (int j = 0; j < 2; ++j) {
      const uint4* p = (const uint4*)(&Bl[buf][(wn * 32 + j * 16 + mrow) * LDS_STRIDE + kbB]);
      ((uint4*)&b[j])[0] = p[0];   // K = kb .. kb+7
      ((uint4*)&b[j])[1] = p[2];   // K = kb+16 .. kb+23
    }
    #pragma unroll
    for (int i = 0; i < 4; ++i)
      #pragma unroll
      for (int j = 0; j < 2; ++j)
        acc[i][j] = __builtin_amdgcn_wmma_f32_16x16x32_f16(
            false, a[i], false, b[j], (short)0, acc[i][j], false, false);
  };

  // ---- prologue: stage tile 0
#if USE_TDM
  if (wv == 0) tdmA(0, 0);
#else
  stageA(0, 0);
#endif
  stageB(0, 0);
#if USE_TDM
  if (wv == 0) __builtin_amdgcn_s_wait_tensorcnt(0);
#endif
  __syncthreads();

  for (int kt = 0; kt < KT; ++kt) {
    if (kt + 2 < KT)  // L2 prefetch of the A tile two steps ahead (global_prefetch_b8)
      __builtin_prefetch(w + (size_t)(cout0 + (tid >> 1)) * Kdim + ((kt + 2) << 5), 0, 1);
    if (kt + 1 < KT) {
#if USE_TDM
      if (wv == 0) tdmA(kt + 1, (kt + 1) & 1);
#else
      stageA(kt + 1, (kt + 1) & 1);
#endif
      stageB(kt + 1, (kt + 1) & 1);
    }
    compute(kt & 1);
#if USE_TDM
    if (wv == 0) __builtin_amdgcn_s_wait_tensorcnt(0);
#endif
    __syncthreads();
  }

  // epilogue: C/D layout -> lane n = lane&15, M = (lane<16 ? r : 8+r)
  const int ncol  = lane & 15;
  const int mbase = (lane < 16) ? 0 : 8;
  #pragma unroll
  for (int i = 0; i < 4; ++i)
    #pragma unroll
    for (int j = 0; j < 2; ++j)
      #pragma unroll
      for (int r = 0; r < 8; ++r) {
        int ch  = cout0 + wm * 64 + i * 16 + mbase + r;
        int pix = p0 + wn * 32 + j * 16 + ncol;
        float v = acc[i][j][r] + bias[ch];
        size_t idx = ((size_t)img * Cout + ch) * 4096 + pix;
        if constexpr (F16OUT) outH[idx] = (_Float16)v;
        else                  outF[idx] = v;
      }
}

// -------------------------------------------------------------- attention part
// qkv: (N, 3072, 4096) f16, ch = [Q: h*256+e][K: 1024+...][V: 2048+...]
__global__ __launch_bounds__(256) void attn_kv_kernel(const _Float16* __restrict__ qkv,
                                                      float* __restrict__ kvOut) {
  const int e = blockIdx.x, n = blockIdx.y;
  const int tid = threadIdx.x, lane = tid & 31, wv = tid >> 5;
  float acc[16];
  #pragma unroll
  for (int t = 0; t < 16; ++t) acc[t] = 0.f;
  for (int l = tid; l < 4096; l += 256) {
    float kh[4], vh[4], ex[4];
    #pragma unroll
    for (int h = 0; h < 4; ++h)
      kh[h] = (float)qkv[((size_t)(n * 3072 + 1024 + h * 256 + e)) * 4096 + l];
    float m = fmaxf(fmaxf(kh[0], kh[1]), fmaxf(kh[2], kh[3]));
    float s = 0.f;
    #pragma unroll
    for (int h = 0; h < 4; ++h) { ex[h] = __expf(kh[h] - m); s += ex[h]; }
    float inv = 1.f / s;
    #pragma unroll
    for (int h = 0; h < 4; ++h)
      vh[h] = (float)qkv[((size_t)(n * 3072 + 2048 + h * 256 + e)) * 4096 + l];
    #pragma unroll
    for (int i = 0; i < 4; ++i)
      #pragma unroll
      for (int j = 0; j < 4; ++j) acc[i * 4 + j] += vh[i] * (ex[j] * inv);
  }
  __shared__ float red[8][16];
  #pragma unroll
  for (int t = 0; t < 16; ++t) acc[t] = wave_sum(acc[t]);
  if (lane == 0) {
    #pragma unroll
    for (int t = 0; t < 16; ++t) red[wv][t] = acc[t];
  }
  __syncthreads();
  if (tid < 16) {
    float s = 0.f;
    #pragma unroll
    for (int w = 0; w < 8; ++w) s += red[w][tid];
    kvOut[((size_t)n * 256 + e) * 16 + tid] = s;
  }
}

// Y[h,l] = sum_h' KV[h,h'] * softmax_h'(Q[:,l]); scatter into padded Yc buffer
__global__ __launch_bounds__(256) void attn_y_kernel(const _Float16* __restrict__ qkv,
                                                     const float* __restrict__ kvIn,
                                                     _Float16* __restrict__ ycp) {
  const int e = blockIdx.x, n = blockIdx.y;
  const int tid = threadIdx.x;
  float kv[16];
  #pragma unroll
  for (int t = 0; t < 16; ++t) kv[t] = kvIn[((size_t)n * 256 + e) * 16 + t];
  for (int l = tid; l < 4096; l += 256) {
    float qh[4], ex[4];
    #pragma unroll
    for (int h = 0; h < 4; ++h)
      qh[h] = (float)qkv[((size_t)(n * 3072 + h * 256 + e)) * 4096 + l];
    float m = fmaxf(fmaxf(qh[0], qh[1]), fmaxf(qh[2], qh[3]));
    float s = 0.f;
    #pragma unroll
    for (int h = 0; h < 4; ++h) { ex[h] = __expf(qh[h] - m); s += ex[h]; }
    float inv = 1.f / s;
    const int hi = l >> 6, wi = l & 63;
    #pragma unroll
    for (int h = 0; h < 4; ++h) {
      float y = kv[h * 4 + 0] * ex[0] + kv[h * 4 + 1] * ex[1] +
                kv[h * 4 + 2] * ex[2] + kv[h * 4 + 3] * ex[3];
      y *= inv;
      size_t dst = (((size_t)(n * 1024 + h * 256 + e) * 66) + hi + 1) * 66 + wi + 1;
      ycp[dst] = (_Float16)y;
    }
  }
}

// ------------------------------------------------------------ final norm + add
__global__ void gn_final_kernel(const float* __restrict__ cp, const float* __restrict__ stats,
                                const float* __restrict__ g, const float* __restrict__ b,
                                const float* __restrict__ x, float* __restrict__ out) {
  size_t gid = (size_t)blockIdx.x * blockDim.x + threadIdx.x;
  if (gid >= (size_t)8 * 128 * 4096) return;
  int n = (int)(gid >> 19);
  int c = (int)((gid >> 12) & 127);
  float mean = stats[2 * n + 0] * (1.f / 524288.f);
  float var  = stats[2 * n + 1] * (1.f / 524288.f) - mean * mean;
  float rstd = rsqrtf(var + 1e-5f);
  out[gid] = (cp[gid] - mean) * rstd * g[c] + b[c] + x[gid];
}

// =============================================================== host driver
extern "C" void kernel_launch(void* const* d_in, const int* in_sizes, int n_in,
                              void* d_out, int out_size, void* d_ws, size_t ws_size,
                              hipStream_t stream) {
  (void)in_sizes; (void)n_in; (void)out_size; (void)ws_size;
  const float* x    = (const float*)d_in[0];
  const float* ng   = (const float*)d_in[1];
  const float* nb   = (const float*)d_in[2];
  const float* Wq   = (const float*)d_in[3];
  const float* bq   = (const float*)d_in[4];
  const float* Wk   = (const float*)d_in[5];
  const float* bk   = (const float*)d_in[6];
  const float* Wv   = (const float*)d_in[7];
  const float* bv   = (const float*)d_in[8];
  const float* Wp   = (const float*)d_in[9];
  const float* bp   = (const float*)d_in[10];
  const float* png  = (const float*)d_in[11];
  const float* pnb  = (const float*)d_in[12];

  char* ws = (char*)d_ws;
  size_t off = 0;
  auto alloc = [&](size_t bytes) { size_t r = off; off = (off + bytes + 255) & ~(size_t)255; return r; };
  const size_t oStatsX = alloc(64);
  const size_t oStatsP = alloc(64);
  const size_t oXnPad  = alloc((size_t)8 * 128 * 66 * 66 * 2);    // 8.9 MB f16
  const size_t oYcp    = alloc((size_t)8 * 1024 * 66 * 66 * 2);   // 71.4 MB f16
  const size_t zeroBytes = off;                                   // zero everything above
  const size_t oWqkv   = alloc((size_t)3072 * 1152 * 2);
  const size_t oBqkv   = alloc((size_t)3072 * 4);
  const size_t oWpk    = alloc((size_t)128 * 9216 * 2);
  const size_t oQkv    = alloc((size_t)8 * 3072 * 4096 * 2);      // 201 MB f16
  const size_t oKv     = alloc((size_t)2048 * 16 * 4);
  const size_t oCpOut  = alloc((size_t)8 * 128 * 4096 * 4);       // 16.8 MB f32

  float*    statsX = (float*)(ws + oStatsX);
  float*    statsP = (float*)(ws + oStatsP);
  _Float16* xnPad  = (_Float16*)(ws + oXnPad);
  _Float16* ycp    = (_Float16*)(ws + oYcp);
  _Float16* wqkv   = (_Float16*)(ws + oWqkv);
  float*    bqkv   = (float*)(ws + oBqkv);
  _Float16* wpk    = (_Float16*)(ws + oWpk);
  _Float16* qkv    = (_Float16*)(ws + oQkv);
  float*    kvb    = (float*)(ws + oKv);
  float*    cpo    = (float*)(ws + oCpOut);

  // 1) zero stats + padded buffers (borders must be exactly 0 every call)
  zero_ws_kernel<<<4096, 256, 0, stream>>>((unsigned int*)ws, zeroBytes / 4);
  // 2) weight/bias packing (f32 -> f16, k = q*Cin + c order)
  pack_qkv_kernel<<<13824, 256, 0, stream>>>(Wq, Wk, Wv, bq, bk, bv, wqkv, bqkv);
  pack_p_kernel<<<4608, 256, 0, stream>>>(Wp, wpk);
  // 3) GroupNorm(1) of x -> padded f16 input
  gn_stats_kernel<<<dim3(64, 8), 256, 0, stream>>>(x, statsX);
  gn_apply_pad_kernel<<<16384, 256, 0, stream>>>(x, statsX, ng, nb, xnPad);
  // 4) fused Q|K|V conv as one WMMA implicit GEMM (Cout=3072, K=1152)
  conv3x3_wmma_kernel<true><<<dim3(24, 32, 8), 256, 0, stream>>>(
      xnPad, wqkv, bqkv, qkv, nullptr, 128, 3072, 1152);
  // 5) attention (tiny): KV = V @ softmax(K), then Y = KV @ softmax(Q) -> padded Yc
  attn_kv_kernel<<<dim3(256, 8), 256, 0, stream>>>(qkv, kvb);
  attn_y_kernel<<<dim3(256, 8), 256, 0, stream>>>(qkv, kvb, ycp);
  // 6) projection conv (Cout=128, K=9216), f32 out
  conv3x3_wmma_kernel<false><<<dim3(1, 32, 8), 256, 0, stream>>>(
      ycp, wpk, bp, nullptr, cpo, 1024, 128, 9216);
  // 7) final GroupNorm + residual
  gn_stats_kernel<<<dim3(64, 8), 256, 0, stream>>>(cpo, statsP);
  gn_final_kernel<<<16384, 256, 0, stream>>>(cpo, statsP, png, pnb, x, (float*)d_out);
}
